// RNN_79164837199890
// MI455X (gfx1250) — matrix operand order repaired and checked by reading, not compiled, not run on hardware
//
#include <hip/hip_runtime.h>
#include <hip/hip_bf16.h>
#include <math.h>

typedef __attribute__((ext_vector_type(16))) __bf16 v16bf;
typedef __attribute__((ext_vector_type(8)))  float  v8f;
typedef __attribute__((ext_vector_type(4)))  unsigned int u32x4;

#define B_SZ  128
#define T_SZ  512
#define FIN   361
#define HID   128
#define NP    362      // policy classes
#define PPAD  368      // 23 * 16
#define K0    512      // padded [x(384) ; h0(128)]
#define K1    256      // [h0 ; h1]
#define N0    512      // 4*H gate columns

union FragBF { v16bf v; u32x4 q[2]; };
union AccF   { v8f   v; float f[8]; };
union PkBF   { __bf16 h[2]; unsigned u; };

__device__ __forceinline__ float sigmoidf_(float x) {
  return 1.0f / (1.0f + __expf(-x));
}

// Opaque zero: always 0, but the compiler must assume it changes each time the
// asm executes. Adding it to the per-timestep weight base pointers makes the
// (t-invariant) fragment load addresses non-loop-invariant, so LLVM cannot
// hoist ~768 VGPRs of weight fragments out of the sequence loop and spill them
// to scratch (round-1 failure). Unlike volatile (round-2 failure: serialized
// SCOPE_SYS flat loads), the loads stay ordinary clause-batched
// global_load_b128 / ds_load_b128 with default RT cache policy.
__device__ __forceinline__ int opaque_zero() {
  int z = 0;
  asm volatile("" : "+v"(z));
  return z;
}

// ---------------------------------------------------------------------------
__global__ void rnn_zero_accum(float* acc) {
  if (threadIdx.x < 8) acc[threadIdx.x] = 0.0f;
}

// ---------------------------------------------------------------------------
// Convert weights fp32 -> bf16, transposed to N-major [n][k] so the WMMA
// B-fragment (lane = column n, 16B K runs at k = 8h and 16 + 8h in lane-half h) is a
// contiguous 2x16B load. W0 K padded 361->384 with zero rows; h weights at
// K = 384..511. Wp padded N 362->368 with zero columns.
// ---------------------------------------------------------------------------
__global__ void rnn_convert_weights(const float* __restrict__ W0,
                                    const float* __restrict__ W1,
                                    const float* __restrict__ Wp,
                                    __bf16* __restrict__ W0t,
                                    __bf16* __restrict__ W1t,
                                    __bf16* __restrict__ Wpt) {
  int id = blockIdx.x * blockDim.x + threadIdx.x;
  const int t0 = N0 * K0;       // 262144
  const int t1 = N0 * K1;       // 131072
  const int t2 = PPAD * HID;    //  47104
  if (id < t0) {
    int n = id / K0, k = id % K0;
    float v = 0.0f;
    if (k < FIN)       v = W0[k * N0 + n];
    else if (k >= 384) v = W0[(FIN + (k - 384)) * N0 + n];
    W0t[id] = (__bf16)v;
  } else if ((id -= t0) < t1) {
    int n = id / K1, k = id % K1;
    W1t[id] = (__bf16)W1[k * N0 + n];
  } else if ((id -= t1) < t2) {
    int n = id / HID, k = id % HID;
    Wpt[id] = (__bf16)((n < NP) ? Wp[k * NP + n] : 0.0f);
  }
}

// ---------------------------------------------------------------------------
// Persistent 2-layer LSTM scan. 8 workgroups x 256 threads (8 waves).
// Each workgroup owns a 16-row batch tile for all 512 timesteps.
// Wave w owns hidden tile w (16 h-columns) = gate tiles {i,f,g,o} at
// N = w*16 + g*128. Cell state c stays in VGPRs for the whole sequence.
// W1^T (256 KB) lives in dynamic LDS; W0^T streams from L2 every step.
// All fragment addresses are (base pointer + constant) so offsets fold into
// VMEM/DS instruction offset immediates.
// ---------------------------------------------------------------------------
__global__ __launch_bounds__(256, 1)
void rnn_lstm_scan(const float* __restrict__ states,
                   const float* __restrict__ b0,
                   const float* __restrict__ b1,
                   const __bf16* __restrict__ W0t,
                   const __bf16* __restrict__ W1t,
                   __bf16* __restrict__ hs) {
  __shared__ __align__(16) __bf16 xh0[16 * K0];   // [row][ x(384) | h0(128) ]
  __shared__ __align__(16) __bf16 h01[16 * K1];   // [row][ h0(128) | h1(128) ]
  extern __shared__ __align__(16) __bf16 w1s[];   // [N0=512][K1=256] bf16

  const int tid   = threadIdx.x;
  const int wave  = tid >> 5;
  const int lane  = tid & 31;
  const int l     = lane & 15;
  const int hi    = lane >> 4;
  const int hcol  = wave * 16 + l;                // hidden column owned by lane
  const int brow0 = blockIdx.x * 16;              // batch-tile base row

  // Preload W1^T into LDS (one-time, 256 KB).
  {
    const u32x4* src = (const u32x4*)W1t;
    u32x4*       dst = (u32x4*)w1s;
    for (int i = tid; i < (N0 * K1) / 8; i += 256) dst[i] = src[i];
  }
  for (int i = tid; i < 16 * K0; i += 256) xh0[i] = (__bf16)0.0f;
  for (int i = tid; i < 16 * K1; i += 256) h01[i] = (__bf16)0.0f;

  float c0v[8], c1v[8];
#pragma unroll
  for (int r = 0; r < 8; ++r) { c0v[r] = 0.0f; c1v[r] = 0.0f; }

  const float b0i = b0[hcol],           b0f = b0[HID + hcol];
  const float b0g = b0[2 * HID + hcol], b0o = b0[3 * HID + hcol];
  const float b1i = b1[hcol],           b1f = b1[HID + hcol];
  const float b1g = b1[2 * HID + hcol], b1o = b1[3 * HID + hcol];

  // Loop-invariant per-lane bases (offsets below are compile-time constants).
  const __bf16* w0base = W0t + (size_t)hcol * K0 + hi * 8;   // B frags, layer0
  const __bf16* a0base = &xh0[l * K0 + hi * 8];              // A frags, layer0
  const __bf16* a1base = &h01[l * K1 + hi * 8];              // A frags, layer1
  const int     w1base = hcol * K1 + hi * 8;                 // B frags, layer1
  const float*  sbase  = states + (size_t)brow0 * T_SZ * FIN;
  size_t hsoff[8];
#pragma unroll
  for (int r = 0; r < 8; ++r)
    hsoff[r] = (size_t)(brow0 + r + 8 * hi) * T_SZ * HID + hcol;

  __syncthreads();

  for (int t = 0; t < T_SZ; ++t) {
    // Per-timestep opaque bases: defeats LICM of the weight fragments.
    const int zk = opaque_zero();
    const __bf16* w0p = w0base + zk;
    const __bf16* w1p = &w1s[w1base + zk];

    // ---- stage x_t (fp32 -> packed bf16) into LDS ------------------------
    {
      const float* st = sbase + (size_t)t * FIN;
#pragma unroll 4
      for (int row = 0; row < 16; ++row) {
        const float* sp = st + (size_t)row * (T_SZ * FIN);
        __bf16*      dp = &xh0[row * K0];
        if (tid < 180) {           // 180 packed pairs cover f = 0..359
          PkBF pk;
          pk.h[0] = (__bf16)sp[2 * tid];
          pk.h[1] = (__bf16)sp[2 * tid + 1];
          *(unsigned*)(dp + 2 * tid) = pk.u;
        } else if (tid == 180) {   // tail element f = 360
          dp[360] = (__bf16)sp[360];
        }
      }
    }
    __syncthreads();

    // ---- layer 0: z0 = [x ; h0] @ W0  (16x512 @ 512x512 slice per wave) --
    AccF a[4];
#pragma unroll
    for (int g = 0; g < 4; ++g)
#pragma unroll
      for (int r = 0; r < 8; ++r) a[g].f[r] = 0.0f;

    for (int kc = 0; kc < K0 / 32; ++kc) {
      FragBF fa;  // A: lane-half K split {0-7,16-23} / {8-15,24-31}
      const __bf16* ap = a0base + kc * 32;
      fa.q[0] = *(const u32x4*)ap;
      fa.q[1] = *(const u32x4*)(ap + 16);
      if (kc + 1 < K0 / 32)  // pull next W0^T K-chunk toward the near cache
        __builtin_prefetch(w0p + (kc + 1) * 32, 0, 3);
#pragma unroll
      for (int g = 0; g < 4; ++g) {
        const __bf16* wp = w0p + (g * (HID * K0) + kc * 32);
        FragBF fb;  // B: lane-half K split {0-7,16-23} / {8-15,24-31}
        fb.q[0] = *(const u32x4*)wp;
        fb.q[1] = *(const u32x4*)(wp + 16);
        a[g].v = __builtin_amdgcn_wmma_f32_16x16x32_bf16(
            false, fa.v, false, fb.v, (short)0, a[g].v, false, false);
      }
    }

    __bf16 h0b[8];
#pragma unroll
    for (int r = 0; r < 8; ++r) {
      float ig = sigmoidf_(a[0].f[r] + b0i);
      float fg = sigmoidf_(a[1].f[r] + b0f + 1.0f);  // forget bias 1.0
      float gg = tanhf(a[2].f[r] + b0g);
      float og = sigmoidf_(a[3].f[r] + b0o);
      float c  = fg * c0v[r] + ig * gg;
      c0v[r]   = c;
      h0b[r]   = (__bf16)(og * tanhf(c));
    }
    __syncthreads();  // all waves done reading old h0 before overwrite
#pragma unroll
    for (int r = 0; r < 8; ++r) {
      int m = r + 8 * hi;  // C-tile: element r <-> batch row M = r + 8*hi
      xh0[m * K0 + 384 + hcol] = h0b[r];
      h01[m * K1 + hcol]       = h0b[r];
    }
    __syncthreads();

    // ---- layer 1: z1 = [h0 ; h1] @ W1  (W1^T from LDS) -------------------
#pragma unroll
    for (int g = 0; g < 4; ++g)
#pragma unroll
      for (int r = 0; r < 8; ++r) a[g].f[r] = 0.0f;

    for (int kc = 0; kc < K1 / 32; ++kc) {
      FragBF fa;
      const __bf16* ap = a1base + kc * 32;
      fa.q[0] = *(const u32x4*)ap;
      fa.q[1] = *(const u32x4*)(ap + 16);
#pragma unroll
      for (int g = 0; g < 4; ++g) {
        const __bf16* wp = w1p + (g * (HID * K1) + kc * 32);
        FragBF fb;
        fb.q[0] = *(const u32x4*)wp;
        fb.q[1] = *(const u32x4*)(wp + 16);
        a[g].v = __builtin_amdgcn_wmma_f32_16x16x32_bf16(
            false, fa.v, false, fb.v, (short)0, a[g].v, false, false);
      }
    }

    __bf16 h1b[8];
#pragma unroll
    for (int r = 0; r < 8; ++r) {
      float ig = sigmoidf_(a[0].f[r] + b1i);
      float fg = sigmoidf_(a[1].f[r] + b1f + 1.0f);
      float gg = tanhf(a[2].f[r] + b1g);
      float og = sigmoidf_(a[3].f[r] + b1o);
      float c  = fg * c1v[r] + ig * gg;
      c1v[r]   = c;
      h1b[r]   = (__bf16)(og * tanhf(c));
    }
    __syncthreads();  // all waves done reading old h1 before overwrite
    {
      const size_t tH = (size_t)t * HID;
#pragma unroll
      for (int r = 0; r < 8; ++r) {
        int m = r + 8 * hi;
        h01[m * K1 + HID + hcol] = h1b[r];
        hs[hsoff[r] + tH]        = h1b[r];
      }
    }
    __syncthreads();
  }
}

// ---------------------------------------------------------------------------
// Heads + losses: per 16-row tile, WMMA policy GEMM [16,128]@[128,368],
// then in-LDS log-softmax / argmax / value head / masked accumulation.
// ---------------------------------------------------------------------------
__global__ __launch_bounds__(128, 2)
void rnn_heads(const __bf16* __restrict__ hs,
               const __bf16* __restrict__ Wpt,
               const float* __restrict__ bpv,
               const float* __restrict__ Wv,
               const float* __restrict__ bv,
               const int* __restrict__ moves,
               const float* __restrict__ values,
               float* __restrict__ accum) {
  __shared__ __align__(16) __bf16 atile[16 * HID];
  __shared__ float logits[16 * PPAD];

  const int tid   = threadIdx.x;
  const int wave  = tid >> 5;
  const int lane  = tid & 31;
  const int l     = lane & 15;
  const int hi    = lane >> 4;
  const int flat0 = blockIdx.x * 16;  // flat row index into [B*T]

  {
    const u32x4* src = (const u32x4*)(hs + (size_t)flat0 * HID);
    u32x4*       dst = (u32x4*)atile;
    for (int i = tid; i < (16 * HID) / 8; i += 128) dst[i] = src[i];
  }
  __syncthreads();

  for (int nt = wave; nt < PPAD / 16; nt += 4) {   // 23 N-tiles over 4 waves
    AccF acc;
#pragma unroll
    for (int r = 0; r < 8; ++r) acc.f[r] = 0.0f;
#pragma unroll
    for (int kc = 0; kc < HID / 32; ++kc) {
      FragBF fa;
      const __bf16* ap = &atile[l * HID + kc * 32 + hi * 8];
      fa.q[0] = *(const u32x4*)ap;
      fa.q[1] = *(const u32x4*)(ap + 16);
      FragBF fb;
      const __bf16* wp = &Wpt[(size_t)(nt * 16 + l) * HID + kc * 32 + hi * 8];
      fb.q[0] = *(const u32x4*)wp;
      fb.q[1] = *(const u32x4*)(wp + 16);
      acc.v = __builtin_amdgcn_wmma_f32_16x16x32_bf16(
          false, fa.v, false, fb.v, (short)0, acc.v, false, false);
    }
    int col  = nt * 16 + l;
    float bb = (col < NP) ? bpv[col] : 0.0f;
#pragma unroll
    for (int r = 0; r < 8; ++r) {
      int m = r + 8 * hi;
      logits[m * PPAD + col] = (col < NP) ? (acc.f[r] + bb) : -1e30f;
    }
  }
  __syncthreads();

  if (tid < 16) {
    const int row = tid;
    float mx = -1e30f; int am = 0;
    for (int p = 0; p < NP; ++p) {
      float v = logits[row * PPAD + p];
      if (v > mx) { mx = v; am = p; }
    }
    float se = 0.0f;
    for (int p = 0; p < NP; ++p) se += __expf(logits[row * PPAD + p] - mx);
    int   flat = flat0 + row;
    int   mv   = moves[flat];
    float xe   = -(logits[row * PPAD + mv] - mx - __logf(se));
    float dot  = 0.0f;
    for (int k = 0; k < HID; ++k) dot += (float)atile[row * HID + k] * Wv[k];
    float w   = tanhf(dot + bv[0]);
    float val = values[flat];
    float msk = (val != -9.0f) ? 1.0f : 0.0f;
    atomicAdd(&accum[0], msk * xe);                     // policy xent sum
    atomicAdd(&accum[1], msk * (w - val) * (w - val));  // value sq-err sum
    atomicAdd(&accum[2], (mv == am) ? 1.0f : 0.0f);     // accuracy count
    atomicAdd(&accum[3], msk);                          // n_mask
  }
}

__global__ void rnn_finalize(const float* __restrict__ accum,
                             float* __restrict__ out) {
  if (threadIdx.x == 0) {
    float n = accum[3];
    out[0] = accum[0] / n;
    out[1] = accum[1] / n;
    out[2] = accum[2] / (float)(B_SZ * T_SZ);
  }
}

// ---------------------------------------------------------------------------
extern "C" void kernel_launch(void* const* d_in, const int* in_sizes, int n_in,
                              void* d_out, int out_size, void* d_ws,
                              size_t ws_size, hipStream_t stream) {
  (void)in_sizes; (void)n_in; (void)out_size; (void)ws_size;
  const float* states = (const float*)d_in[0];
  const int*   moves  = (const int*)d_in[1];
  const float* values = (const float*)d_in[2];
  const float* W0     = (const float*)d_in[3];
  const float* b0     = (const float*)d_in[4];
  const float* W1     = (const float*)d_in[5];
  const float* b1     = (const float*)d_in[6];
  const float* Wp     = (const float*)d_in[7];
  const float* bp     = (const float*)d_in[8];
  const float* Wv     = (const float*)d_in[9];
  const float* bv     = (const float*)d_in[10];
  float* out = (float*)d_out;

  // Workspace layout (all 16B-aligned): accum | W0t | W1t | Wpt | hs
  char*   ws    = (char*)d_ws;
  float*  accum = (float*)ws;
  __bf16* W0t   = (__bf16*)(ws + 256);
  __bf16* W1t   = (__bf16*)(ws + 256 + (size_t)N0 * K0 * 2);
  __bf16* Wpt   = (__bf16*)(ws + 256 + (size_t)N0 * K0 * 2 + (size_t)N0 * K1 * 2);
  __bf16* hsbuf = (__bf16*)(ws + 256 + (size_t)N0 * K0 * 2 + (size_t)N0 * K1 * 2 +
                            (size_t)PPAD * HID * 2);

  rnn_zero_accum<<<1, 32, 0, stream>>>(accum);

  {
    int total = N0 * K0 + N0 * K1 + PPAD * HID;  // 440320 = 1720 * 256
    rnn_convert_weights<<<(total + 255) / 256, 256, 0, stream>>>(
        W0, W1, Wp, W0t, W1t, Wpt);
  }

  // 8 persistent workgroups (one 16-row batch tile each); 256 KB dynamic LDS
  // for resident W1^T (+24 KB static state buffers = 280 KB of the 320 KB WGP).
  rnn_lstm_scan<<<8, 256, (size_t)N0 * K1 * 2, stream>>>(
      states, b0, b1, W0t, W1t, hsbuf);

  rnn_heads<<<(B_SZ * T_SZ) / 16, 128, 0, stream>>>(
      hsbuf, Wpt, bp, Wv, bv, moves, values, accum);

  rnn_finalize<<<1, 32, 0, stream>>>(accum, out);
}
